// GraphMambaModel_72791105732715
// MI455X (gfx1250) — compile-verified, matrix operand order checked by reading, not run on hardware
//
#include <hip/hip_runtime.h>
#include <hip/hip_bf16.h>

// ---------------------------------------------------------------------------
// GraphMamba forward for MI455X (gfx1250, wave32, WMMA).
// Input leaf enumeration (setup_inputs() insertion order, recursive):
//  0: x | Fire 1-6 | Terrain 7-12 | Burn 13-16 | Soil 17-22
//  RainAcc 23-37 (ln_g,ln_b,proj_w,proj_b, w_in,conv_w,conv_b,w_x,w_dt,b_dt,A_log,D,w_out, on_g,on_b)
//  RainInt 38-52 | Storm 53-58 | fusion 59-60 | spectral 61-62
//  sm1 63-71 | sn1 72-73 | sm2 74-82 | sn2 83-84 | cls 85-90
// ---------------------------------------------------------------------------

typedef __attribute__((ext_vector_type(16))) _Float16 v16h;
typedef __attribute__((ext_vector_type(8)))  _Float16 v8h;
typedef __attribute__((ext_vector_type(8)))  float    v8f;

#define NTOK   16384      // B*K
#define BATCH  2048
#define SEQ    8
#define DI     128
#define PFP    192        // padded pf pitch (168 data + 24 zero)
#define XDG    48         // padded group x_dbl pitch (36 data)
#define XDS    272        // padded sm x_dbl pitch (260 data)

struct Ptrs { const float* p[96]; };

__device__ __forceinline__ float siluf(float x)     { return x / (1.f + expf(-x)); }
__device__ __forceinline__ float softplusf(float x) { return (x > 20.f) ? x : log1pf(expf(x)); }

// ---------------------------------------------------------------------------
// Guard-free WMMA GEMM: C[M, Npad] = act(A[M,K](f32) * Wh[Npad,K](f16)^T + bias)
// Host guarantees: M%16==0, K%32==0, Wh zero-padded fp16, 16B alignment.
// One wave32 computes a 16 x (16*NT) strip; A fragment reused for NT WMMAs.
// ---------------------------------------------------------------------------
__device__ __forceinline__ v16h fragA(const float* __restrict__ base, int k0, int hb)
{
    const float4* q0 = reinterpret_cast<const float4*>(base + k0 + hb * 8);
    const float4* q1 = reinterpret_cast<const float4*>(base + k0 + 16 + hb * 8);
    float4 f0 = q0[0], f1 = q0[1], f2 = q1[0], f3 = q1[1];
    v16h a;
    a[0]  = (_Float16)f0.x; a[1]  = (_Float16)f0.y; a[2]  = (_Float16)f0.z; a[3]  = (_Float16)f0.w;
    a[4]  = (_Float16)f1.x; a[5]  = (_Float16)f1.y; a[6]  = (_Float16)f1.z; a[7]  = (_Float16)f1.w;
    a[8]  = (_Float16)f2.x; a[9]  = (_Float16)f2.y; a[10] = (_Float16)f2.z; a[11] = (_Float16)f2.w;
    a[12] = (_Float16)f3.x; a[13] = (_Float16)f3.y; a[14] = (_Float16)f3.z; a[15] = (_Float16)f3.w;
    return a;
}

__device__ __forceinline__ v16h fragB(const _Float16* __restrict__ base, int k0, int hb)
{
    v8h lo = *reinterpret_cast<const v8h*>(base + k0 + hb * 8);
    v8h hi = *reinterpret_cast<const v8h*>(base + k0 + 16 + hb * 8);
    v16h b;
#pragma unroll
    for (int i = 0; i < 8; ++i) { b[i] = lo[i]; b[8 + i] = hi[i]; }
    return b;
}

template<int NT>
__global__ void wmma_gemm_nt(const float* __restrict__ A, int lda,
                             const _Float16* __restrict__ W, int ldw,
                             const float* __restrict__ bias,
                             float* __restrict__ C, int ldc, int K, int act)
{
    const int m0   = blockIdx.x * 16;
    const int n0   = blockIdx.y * 16 * NT;
    const int lane = threadIdx.x & 31;
    const int r    = lane & 15;
    const int hb   = lane >> 4;

    const float* Ap = A + (size_t)(m0 + r) * lda;
    const _Float16* Wp[NT];
#pragma unroll
    for (int t = 0; t < NT; ++t) Wp[t] = W + (size_t)(n0 + t * 16 + r) * ldw;

    v8f acc[NT];
#pragma unroll
    for (int t = 0; t < NT; ++t) acc[t] = (v8f){0.f,0.f,0.f,0.f,0.f,0.f,0.f,0.f};

    int k0 = 0;
    do {
        v16h a = fragA(Ap, k0, hb);
#pragma unroll
        for (int t = 0; t < NT; ++t) {
            v16h b = fragB(Wp[t], k0, hb);
            acc[t] = __builtin_amdgcn_wmma_f32_16x16x32_f16(false, a, false, b,
                                                            (short)0, acc[t], false, false);
        }
        k0 += 32;
    } while (k0 < K);

#pragma unroll
    for (int t = 0; t < NT; ++t) {
        int n = n0 + t * 16 + r;
        float bv = bias ? bias[n] : 0.f;
#pragma unroll
        for (int v = 0; v < 8; ++v) {
            float rv = acc[t][v] + bv;
            if (act) rv = fmaxf(rv, 0.f);
            C[(size_t)(m0 + v + 8 * hb) * ldc + n] = rv;
        }
    }
}

// zero-pad + fp32->fp16 convert a weight matrix src(N,K) -> dst(Npad,Kpad)
__global__ void k_pad_wh(const float* __restrict__ src, _Float16* __restrict__ dst,
                         int N, int K, int Npad, int Kpad)
{
    int id = blockIdx.x * blockDim.x + threadIdx.x;
    if (id >= Npad * Kpad) return;
    int n = id / Kpad, k = id % Kpad;
    dst[id] = (_Float16)((n < N && k < K) ? src[n * K + k] : 0.f);
}

// zero pf pad columns [168,192)
__global__ void k_zero_pfpad(float* __restrict__ pf)
{
    int id = blockIdx.x * blockDim.x + threadIdx.x;
    if (id >= NTOK * (PFP - 168)) return;
    int t = id / (PFP - 168), c = id % (PFP - 168);
    pf[(size_t)t * PFP + 168 + c] = 0.f;
}

// ---------------------------------------------------------------------------
// Per-token feature groups + Rain LN+projections.
// ---------------------------------------------------------------------------
__device__ __forceinline__ void ln2(float a0, float a1, const float* g, const float* b,
                                    float& n0, float& n1)
{
    float m  = 0.5f * (a0 + a1);
    float d0 = a0 - m, d1 = a1 - m;
    float v  = 0.5f * (d0 * d0 + d1 * d1);
    float rs = rsqrtf(v + 1e-5f);
    n0 = d0 * rs * g[0] + b[0];
    n1 = d1 * rs * g[1] + b[1];
}

__device__ __forceinline__ void mlp2(float a0, float a1,
                                     const float* g, const float* b,
                                     const float* w1, const float* b1,
                                     const float* w2, const float* b2,
                                     float* out)
{
    float n0, n1;
    ln2(a0, a1, g, b, n0, n1);
    float h[4];
#pragma unroll
    for (int j = 0; j < 4; ++j)
        h[j] = fmaxf(w1[j * 2] * n0 + w1[j * 2 + 1] * n1 + b1[j], 0.f);
#pragma unroll
    for (int o = 0; o < 8; ++o) {
        float s = b2[o];
#pragma unroll
        for (int j = 0; j < 4; ++j) s += w2[o * 4 + j] * h[j];
        out[o] = s;
    }
}

__device__ __forceinline__ void proj4(const float* a,
                                      const float* g, const float* b,
                                      const float* pw, const float* pb,
                                      float* out)
{
    float m = 0.25f * (a[0] + a[1] + a[2] + a[3]);
    float d[4]; float v = 0.f;
#pragma unroll
    for (int i = 0; i < 4; ++i) { d[i] = a[i] - m; v += d[i] * d[i]; }
    v *= 0.25f;
    float rs = rsqrtf(v + 1e-5f);
    float n[4];
#pragma unroll
    for (int i = 0; i < 4; ++i) n[i] = d[i] * rs * g[i] + b[i];
    for (int o = 0; o < 64; ++o) {
        float s = pb[o];
#pragma unroll
        for (int i = 0; i < 4; ++i) s += pw[o * 4 + i] * n[i];
        out[o] = s;
    }
}

__global__ void k_groups(Ptrs P, const float* __restrict__ x,
                         float* __restrict__ pf,
                         float* __restrict__ hA, float* __restrict__ hI)
{
    int t = blockIdx.x * blockDim.x + threadIdx.x;
    if (t >= NTOK) return;
    float xr[17];
#pragma unroll
    for (int i = 0; i < 17; ++i) xr[i] = x[t * 17 + i];
    float* out = pf + (size_t)t * PFP;

    mlp2(xr[0],  xr[1],  P.p[1],  P.p[2],  P.p[3],  P.p[4],  P.p[5],  P.p[6],  out + 0);   // Fire
    mlp2(xr[2],  xr[3],  P.p[7],  P.p[8],  P.p[9],  P.p[10], P.p[11], P.p[12], out + 8);   // Terrain
    {                                                                                      // Burn [4,2]
        float n0, n1;
        ln2(xr[4], xr[2], P.p[13], P.p[14], n0, n1);
#pragma unroll
        for (int o = 0; o < 8; ++o)
            out[16 + o] = P.p[16][o] + P.p[15][o * 2] * n0 + P.p[15][o * 2 + 1] * n1;
    }
    mlp2(xr[5],  xr[6],  P.p[17], P.p[18], P.p[19], P.p[20], P.p[21], P.p[22], out + 24);  // Soil
    mlp2(xr[15], xr[16], P.p[53], P.p[54], P.p[55], P.p[56], P.p[57], P.p[58], out + 160); // Storm

    proj4(&xr[7],  P.p[23], P.p[24], P.p[25], P.p[26], hA + (size_t)t * 64);               // RainAcc
    proj4(&xr[11], P.p[38], P.p[39], P.p[40], P.p[41], hI + (size_t)t * 64);               // RainInt
}

// ---------------------------------------------------------------------------
// Group-mamba (L=1)
// ---------------------------------------------------------------------------
__global__ void k_gm_u(Ptrs P, int icw, int icb,
                       const float* __restrict__ xz, float* __restrict__ u)
{
    int id = blockIdx.x * blockDim.x + threadIdx.x;
    if (id >= NTOK * DI) return;
    int t = id >> 7, d = id & 127;
    float xi  = xz[(size_t)t * 256 + d];
    float pre = P.p[icw][d * 4 + 3] * xi + P.p[icb][d];
    u[(size_t)t * DI + d] = siluf(pre);
}

// y = (softplus(dt)*u*<B,C> + u*D) * silu(z)
__global__ void k_gm_combine(Ptrs P, int iwdt, int ibdt, int iD,
                             const float* __restrict__ xdbl,
                             const float* __restrict__ xz,
                             const float* __restrict__ u,
                             float* __restrict__ y)
{
    int t = blockIdx.x;
    int d = threadIdx.x;
    const float* xd = xdbl + (size_t)t * XDG;
    float bc = 0.f;
#pragma unroll
    for (int s = 0; s < 16; ++s) bc += xd[4 + s] * xd[20 + s];
    float dtr = P.p[ibdt][d];
#pragma unroll
    for (int j = 0; j < 4; ++j) dtr += P.p[iwdt][d * 4 + j] * xd[j];
    float dt = softplusf(dtr);
    float uu = u[(size_t)t * DI + d];
    float zz = xz[(size_t)t * 256 + 128 + d];
    y[(size_t)t * DI + d] = (dt * uu * bc + uu * P.p[iD][d]) * siluf(zz);
}

// out_group = h + LN(mpre)  ->  pf[:, col:col+64]
__global__ void k_gm_finish(Ptrs P, int iong, int ionb,
                            const float* __restrict__ mpre,
                            const float* __restrict__ h,
                            float* __restrict__ pf, int col)
{
    int t = blockIdx.x * blockDim.x + threadIdx.x;
    if (t >= NTOK) return;
    const float* mp = mpre + (size_t)t * 64;
    float mean = 0.f;
    for (int d = 0; d < 64; ++d) mean += mp[d];
    mean *= (1.f / 64.f);
    float var = 0.f;
    for (int d = 0; d < 64; ++d) { float dd = mp[d] - mean; var += dd * dd; }
    var *= (1.f / 64.f);
    float rs = rsqrtf(var + 1e-5f);
    float* o = pf + (size_t)t * PFP + col;
    for (int d = 0; d < 64; ++d)
        o[d] = h[(size_t)t * 64 + d] + (mp[d] - mean) * rs * P.p[iong][d] + P.p[ionb][d];
}

// ---------------------------------------------------------------------------
// Sequence mamba (L=8)
// ---------------------------------------------------------------------------
__global__ void k_conv(Ptrs P, int icw, int icb,
                       const float* __restrict__ xz, float* __restrict__ u)
{
    int id = blockIdx.x * blockDim.x + threadIdx.x;
    if (id >= BATCH * DI) return;
    int b = id >> 7, d = id & 127;
    const float* cw = P.p[icw] + d * 4;
    float cb = P.p[icb][d];
    float xi[SEQ];
#pragma unroll
    for (int t = 0; t < SEQ; ++t) xi[t] = xz[((size_t)(b * SEQ + t)) * 256 + d];
#pragma unroll
    for (int t = 0; t < SEQ; ++t) {
        float s = cb;
#pragma unroll
        for (int j = 0; j < 4; ++j) {
            int tt = t - 3 + j;
            if (tt >= 0) s += cw[j] * xi[tt];
        }
        u[((size_t)(b * SEQ + t)) * DI + d] = siluf(s);
    }
}

// Selective scan, S=128: one wave per (batch, channel), 4 states per lane.
__global__ void k_scan(Ptrs P, int iwdt, int ibdt, int iAlog, int iD,
                       const float* __restrict__ xdbl,
                       const float* __restrict__ xz,
                       const float* __restrict__ u,
                       float* __restrict__ y)
{
    int wid  = (blockIdx.x * blockDim.x + threadIdx.x) >> 5;
    int lane = threadIdx.x & 31;
    if (wid >= BATCH * DI) return;
    int b = wid >> 7, d = wid & 127;

    float Aq[4], h[4];
#pragma unroll
    for (int q = 0; q < 4; ++q) {
        Aq[q] = -expf(P.p[iAlog][(size_t)d * 128 + lane * 4 + q]);
        h[q]  = 0.f;
    }
    const float* wdt = P.p[iwdt] + d * 4;
    float bdt = P.p[ibdt][d];
    float Dv  = P.p[iD][d];

#pragma unroll
    for (int t = 0; t < SEQ; ++t) {
        size_t row = (size_t)(b * SEQ + t);
        const float* xd = xdbl + row * XDS;
        float dtr = bdt;
#pragma unroll
        for (int j = 0; j < 4; ++j) dtr += wdt[j] * xd[j];
        float dt = softplusf(dtr);
        float ut = u[row * DI + d];
        float du = dt * ut;
        float yacc = 0.f;
#pragma unroll
        for (int q = 0; q < 4; ++q) {
            int s = lane * 4 + q;
            h[q]  = expf(dt * Aq[q]) * h[q] + du * xd[4 + s];
            yacc += h[q] * xd[132 + s];
        }
        for (int off = 16; off > 0; off >>= 1)
            yacc += __shfl_xor(yacc, off, 32);
        if (lane == 0) {
            float zz = xz[row * 256 + 128 + d];
            y[row * DI + d] = (yacc + ut * Dv) * siluf(zz);
        }
    }
}

// out = inp + LN(mpre) * g + b
__global__ void k_resid_ln(Ptrs P, int ig, int ib,
                           const float* __restrict__ inp,
                           const float* __restrict__ mpre,
                           float* __restrict__ out)
{
    int t = blockIdx.x * blockDim.x + threadIdx.x;
    if (t >= NTOK) return;
    const float* mp = mpre + (size_t)t * 64;
    float mean = 0.f;
    for (int d = 0; d < 64; ++d) mean += mp[d];
    mean *= (1.f / 64.f);
    float var = 0.f;
    for (int d = 0; d < 64; ++d) { float dd = mp[d] - mean; var += dd * dd; }
    var *= (1.f / 64.f);
    float rs = rsqrtf(var + 1e-5f);
    for (int d = 0; d < 64; ++d)
        out[(size_t)t * 64 + d] = inp[(size_t)t * 64 + d]
                                + (mp[d] - mean) * rs * P.p[ig][d] + P.p[ib][d];
}

// f = concat(spec, mean_t(s2))
__global__ void k_pool_concat(const float* __restrict__ spec,
                              const float* __restrict__ s2,
                              float* __restrict__ f)
{
    int id = blockIdx.x * blockDim.x + threadIdx.x;
    if (id >= BATCH * 128) return;
    int b = id >> 7, c = id & 127;
    if (c < 64) {
        f[(size_t)b * 128 + c] = spec[(size_t)b * 64 + c];
    } else {
        int d = c - 64;
        float s = 0.f;
#pragma unroll
        for (int t = 0; t < SEQ; ++t) s += s2[((size_t)(b * SEQ + t)) * 64 + d];
        f[(size_t)b * 128 + c] = s * (1.f / SEQ);
    }
}

__global__ void k_final(Ptrs P, const float* __restrict__ h2, float* __restrict__ out)
{
    int b = blockIdx.x * blockDim.x + threadIdx.x;
    if (b >= BATCH) return;
    float s = P.p[90][0];
#pragma unroll
    for (int j = 0; j < 32; ++j) s += P.p[89][j] * h2[(size_t)b * 32 + j];
    out[b] = 1.f / (1.f + expf(-s));
}

// ---------------------------------------------------------------------------
// Host orchestration
// ---------------------------------------------------------------------------
static void gemm(hipStream_t s, const float* A, int lda, const _Float16* W, int ldw,
                 const float* bias, float* C, int ldc, int M, int Npad, int K, int act)
{
    dim3 blk(32);
    if (Npad % 64 == 0) {
        dim3 g(M / 16, Npad / 64);
        hipLaunchKernelGGL(wmma_gemm_nt<4>, g, blk, 0, s, A, lda, W, ldw, bias, C, ldc, K, act);
    } else if (Npad % 32 == 0) {
        dim3 g(M / 16, Npad / 32);
        hipLaunchKernelGGL(wmma_gemm_nt<2>, g, blk, 0, s, A, lda, W, ldw, bias, C, ldc, K, act);
    } else {
        dim3 g(M / 16, Npad / 16);
        hipLaunchKernelGGL(wmma_gemm_nt<1>, g, blk, 0, s, A, lda, W, ldw, bias, C, ldc, K, act);
    }
}

static void pad_wh(hipStream_t s, const float* src, _Float16* dst, int N, int K, int Npad, int Kpad)
{
    int n = Npad * Kpad;
    hipLaunchKernelGGL(k_pad_wh, dim3((n + 255) / 256), dim3(256), 0, s, src, dst, N, K, Npad, Kpad);
}

extern "C" void kernel_launch(void* const* d_in, const int* in_sizes, int n_in,
                              void* d_out, int out_size, void* d_ws, size_t ws_size,
                              hipStream_t stream)
{
    (void)in_sizes; (void)out_size; (void)ws_size;

    Ptrs P;
    for (int i = 0; i < 96; ++i)
        P.p[i] = (i < n_in) ? (const float*)d_in[i] : nullptr;
    const float* x = P.p[0];

    float* ws = (float*)d_ws;
    float* XZ   = ws;                       // 16384*256
    float* XDBL = XZ   + (size_t)NTOK*256;  // 16384*272
    float* UB   = XDBL + (size_t)NTOK*XDS;  // 16384*128
    float* YB   = UB   + (size_t)NTOK*128;  // 16384*128
    float* PF   = YB   + (size_t)NTOK*128;  // 16384*192 (padded pitch)
    float* HA   = PF   + (size_t)NTOK*PFP;  // 16384*64
    float* HI   = HA   + (size_t)NTOK*64;   // 16384*64
    float* MP   = HI   + (size_t)NTOK*64;   // 16384*64
    float* XP   = MP   + (size_t)NTOK*64;   // 16384*64
    float* S1   = XP   + (size_t)NTOK*64;   // 16384*64
    float* SPEC = S1   + (size_t)NTOK*64;   // 2048*64
    float* FB   = SPEC + (size_t)BATCH*64;  // 2048*128
    float* H1   = FB   + (size_t)BATCH*128; // 2048*64
    float* H2   = H1   + (size_t)BATCH*64;  // 2048*32
    float* S2   = HA;                       // reuse (HA dead after group finish)

    // fp16 padded-weight arena (carved after fp32 buffers; all slots 16B-aligned)
    _Float16* WH = (_Float16*)(H2 + (size_t)BATCH * 32);
    _Float16* cur = WH;
    auto carve = [&](int Npad, int Kpad) { _Float16* r = cur; cur += (size_t)Npad * Kpad; return r; };
    _Float16* W_IN[4]; _Float16* W_XG[2]; _Float16* W_XS[2]; _Float16* W_OUT[4];
    const int win_src[4]  = {27, 42, 63, 74};
    const int wout_src[4] = {35, 50, 71, 82};
    const int wxg_src[2]  = {30, 45};
    const int wxs_src[2]  = {66, 77};
    for (int i = 0; i < 4; ++i) W_IN[i]  = carve(256, 64);
    for (int i = 0; i < 4; ++i) W_OUT[i] = carve(64, 128);
    for (int i = 0; i < 2; ++i) W_XG[i]  = carve(XDG, 128);
    for (int i = 0; i < 2; ++i) W_XS[i]  = carve(XDS, 128);
    _Float16* W_FUS  = carve(64, PFP);
    _Float16* W_SPC  = carve(64, 64);
    _Float16* W_CLS1 = carve(64, 128);
    _Float16* W_CLS2 = carve(32, 64);

    // 0) pre-convert all GEMM weights to padded fp16 (deterministic each call)
    for (int i = 0; i < 4; ++i) pad_wh(stream, P.p[win_src[i]],  W_IN[i],  256, 64,  256, 64);
    for (int i = 0; i < 4; ++i) pad_wh(stream, P.p[wout_src[i]], W_OUT[i], 64, 128,  64, 128);
    for (int i = 0; i < 2; ++i) pad_wh(stream, P.p[wxg_src[i]],  W_XG[i],  36, 128,  XDG, 128);
    for (int i = 0; i < 2; ++i) pad_wh(stream, P.p[wxs_src[i]],  W_XS[i],  260, 128, XDS, 128);
    pad_wh(stream, P.p[59], W_FUS,  64, 168, 64, PFP);
    pad_wh(stream, P.p[61], W_SPC,  64, 64,  64, 64);
    pad_wh(stream, P.p[85], W_CLS1, 64, 128, 64, 128);
    pad_wh(stream, P.p[87], W_CLS2, 32, 64,  32, 64);
    hipLaunchKernelGGL(k_zero_pfpad, dim3((NTOK * (PFP - 168) + 255) / 256), dim3(256), 0, stream, PF);

    // 1) per-token groups + rain projections
    hipLaunchKernelGGL(k_groups, dim3((NTOK + 255) / 256), dim3(256), 0, stream,
                       P, x, PF, HA, HI);

    // 2) group mambas (L=1)
    const int gb[2]   = {27, 42};
    const int gon[2]  = {36, 51};
    const int gcol[2] = {32, 96};
    float* gh[2] = {HA, HI};
    for (int g = 0; g < 2; ++g) {
        int b = gb[g];
        gemm(stream, gh[g], 64, W_IN[g], 64, nullptr, XZ, 256, NTOK, 256, 64, 0);     // xz
        hipLaunchKernelGGL(k_gm_u, dim3((NTOK * DI + 255) / 256), dim3(256), 0, stream,
                           P, b + 1, b + 2, XZ, UB);
        gemm(stream, UB, 128, W_XG[g], 128, nullptr, XDBL, XDG, NTOK, XDG, 128, 0);   // x_dbl
        hipLaunchKernelGGL(k_gm_combine, dim3(NTOK), dim3(DI), 0, stream,
                           P, b + 4, b + 5, b + 7, XDBL, XZ, UB, YB);
        gemm(stream, YB, 128, W_OUT[g], 128, nullptr, MP, 64, NTOK, 64, 128, 0);      // w_out
        hipLaunchKernelGGL(k_gm_finish, dim3((NTOK + 255) / 256), dim3(256), 0, stream,
                           P, gon[g], gon[g] + 1, MP, gh[g], PF, gcol[g]);
    }

    // 3) fusion (K padded 168->192; pf pad cols are zero) + spectral
    gemm(stream, PF, PFP, W_FUS, PFP, P.p[60], XP, 64, NTOK, 64, PFP, 1);
    gemm(stream, XP, 512, W_SPC, 64, P.p[62], SPEC, 64, BATCH, 64, 64, 1);            // xp[:,0,:]

    // 4) two sequence mambas
    const int sb[2]  = {63, 74};
    const int sn[2]  = {72, 83};
    const float* sin_[2] = {XP, S1};
    float* sout[2]       = {S1, S2};
    for (int m = 0; m < 2; ++m) {
        int b = sb[m];
        gemm(stream, sin_[m], 64, W_IN[2 + m], 64, nullptr, XZ, 256, NTOK, 256, 64, 0);
        hipLaunchKernelGGL(k_conv, dim3((BATCH * DI + 255) / 256), dim3(256), 0, stream,
                           P, b + 1, b + 2, XZ, UB);
        gemm(stream, UB, 128, W_XS[m], 128, nullptr, XDBL, XDS, NTOK, XDS, 128, 0);
        hipLaunchKernelGGL(k_scan, dim3((BATCH * DI) / 8), dim3(256), 0, stream,
                           P, b + 4, b + 5, b + 6, b + 7, XDBL, XZ, UB, YB);
        gemm(stream, YB, 128, W_OUT[2 + m], 128, nullptr, MP, 64, NTOK, 64, 128, 0);
        hipLaunchKernelGGL(k_resid_ln, dim3((NTOK + 255) / 256), dim3(256), 0, stream,
                           P, sn[m], sn[m] + 1, sin_[m], MP, sout[m]);
    }

    // 5) head
    hipLaunchKernelGGL(k_pool_concat, dim3((BATCH * 128 + 255) / 256), dim3(256), 0, stream,
                       SPEC, S2, FB);
    gemm(stream, FB, 128, W_CLS1, 128, P.p[86], H1, 64, BATCH, 64, 128, 1);
    gemm(stream, H1, 64,  W_CLS2, 64,  P.p[88], H2, 32, BATCH, 32, 64, 1);
    hipLaunchKernelGGL(k_final, dim3((BATCH + 255) / 256), dim3(256), 0, stream,
                       P, H2, (float*)d_out);
}
